// YvMoEGate_83597243449508
// MI455X (gfx1250) — matrix operand order repaired and checked by reading, not compiled, not run on hardware
//
#include <hip/hip_runtime.h>
#include <math.h>

typedef __attribute__((ext_vector_type(16))) _Float16 v16h;
typedef __attribute__((ext_vector_type(4)))  _Float16 v4h;
typedef __attribute__((ext_vector_type(8)))  float    v8f;
typedef __attribute__((ext_vector_type(4)))  float    v4f;

#define NUM_EXPERTS 64
#define HIDDEN      4096
#define TEMP        3.0f
#define LB_ALPHA    0.01f
#define Z_ALPHA     1e-4f
#define LDS_STRIDE  68           // 64 + 4 pad to break bank conflicts
#define GWH_OFFSET  1024         // byte offset of f16 gate_w inside workspace

// ---------------------------------------------------------------------------
// ws layout: bytes [0, 516): reduction floats
//            [0..63] sel-count sums, [64..127] prob sums, [128] z-sum
//            bytes [1024, 1024+E*H*2): gate_w converted to f16
// ---------------------------------------------------------------------------
__global__ void zero_ws_kernel(float* __restrict__ ws) {
    int t = threadIdx.x;
    if (t < 129) ws[t] = 0.0f;
}

// One-shot gate_w fp32 -> f16 conversion (E*H = 262144 elements, L2-resident).
__global__ __launch_bounds__(256) void convert_gw_kernel(
    const float* __restrict__ gw, _Float16* __restrict__ gwh) {
    int i = (blockIdx.x * 256 + threadIdx.x) * 4;
    v4f v = *(const v4f*)(gw + i);
    v4h h;
    h[0] = (_Float16)v[0]; h[1] = (_Float16)v[1];
    h[2] = (_Float16)v[2]; h[3] = (_Float16)v[3];
    *(v4h*)(gwh + i) = h;
}

// ---------------------------------------------------------------------------
// Fused gating kernel: WMMA GEMM (16 tokens x 64 experts per wave) + softmax
// + top-2 + per-expert / z-loss partial reductions.
// Block = 128 threads (4 waves), covers 64 tokens. Grid = T/64.
// ---------------------------------------------------------------------------
__global__ __launch_bounds__(128) void moe_gate_kernel(
    const float*    __restrict__ x,      // [T, H] fp32 (HBM-bound stream)
    const _Float16* __restrict__ gwh,    // [E, H] f16 (pre-converted, L2)
    const float*    __restrict__ bias,   // [E]
    float* __restrict__ out_scores,      // [T, 2]
    int*   __restrict__ out_idx,         // [T, 2]
    float* __restrict__ ws)              // reduction floats
{
    __shared__ float lds_logits[64 * LDS_STRIDE];
    __shared__ float lds_sel[NUM_EXPERTS];
    __shared__ float lds_prob[NUM_EXPERTS];
    __shared__ float lds_z;

    const int tid    = threadIdx.x;
    const int wave   = tid >> 5;
    const int lane   = tid & 31;
    const int lhalf  = lane >> 4;     // 0: lanes 0-15, 1: lanes 16-31
    const int lmod   = lane & 15;
    const int wgTok0 = blockIdx.x * 64;
    const int tokBase = wgTok0 + wave * 16;

    if (tid < NUM_EXPERTS) { lds_sel[tid] = 0.0f; lds_prob[tid] = 0.0f; }
    if (tid == 0) lds_z = 0.0f;

    // ---------------- WMMA GEMM: logits tile [16 tokens x 64 experts] ------
    v8f acc0 = {}, acc1 = {}, acc2 = {}, acc3 = {};

    // A-fragment base: row = token, per-lane K offset {0 or 16}, 16 contiguous K.
    const float* aptr = x + (size_t)(tokBase + lmod) * HIDDEN + (lhalf << 4);
    // B-fragment base (f16): row lmod of expert tile, per-lane K off {0 or 16}.
    const _Float16* bptr = gwh + (size_t)lmod * HIDDEN + (lhalf << 4);

    for (int k0 = 0; k0 < HIDDEN; k0 += 32) {
        // ---- A fragment (16x32 f16): lane holds K {a+0..15} ----
        v4f a0 = *(const v4f*)(aptr + k0);
        v4f a1 = *(const v4f*)(aptr + k0 + 4);
        v4f a2 = *(const v4f*)(aptr + k0 + 8);
        v4f a3 = *(const v4f*)(aptr + k0 + 12);
        v16h A;
        A[0]  = (_Float16)a0[0]; A[1]  = (_Float16)a0[1];
        A[2]  = (_Float16)a0[2]; A[3]  = (_Float16)a0[3];
        A[4]  = (_Float16)a1[0]; A[5]  = (_Float16)a1[1];
        A[6]  = (_Float16)a1[2]; A[7]  = (_Float16)a1[3];
        A[8]  = (_Float16)a2[0]; A[9]  = (_Float16)a2[1];
        A[10] = (_Float16)a2[2]; A[11] = (_Float16)a2[3];
        A[12] = (_Float16)a3[0]; A[13] = (_Float16)a3[1];
        A[14] = (_Float16)a3[2]; A[15] = (_Float16)a3[3];

        // ---- B fragments (32x16 f16): direct packed-f16 loads, no cvt ----
        v16h B0 = *(const v16h*)(bptr + (size_t)0 * 16 * HIDDEN + k0);
        v16h B1 = *(const v16h*)(bptr + (size_t)1 * 16 * HIDDEN + k0);
        v16h B2 = *(const v16h*)(bptr + (size_t)2 * 16 * HIDDEN + k0);
        v16h B3 = *(const v16h*)(bptr + (size_t)3 * 16 * HIDDEN + k0);

        acc0 = __builtin_amdgcn_wmma_f32_16x16x32_f16(false, A, false, B0, (short)0, acc0, false, false);
        acc1 = __builtin_amdgcn_wmma_f32_16x16x32_f16(false, A, false, B1, (short)0, acc1, false, false);
        acc2 = __builtin_amdgcn_wmma_f32_16x16x32_f16(false, A, false, B2, (short)0, acc2, false, false);
        acc3 = __builtin_amdgcn_wmma_f32_16x16x32_f16(false, A, false, B3, (short)0, acc3, false, false);
    }

    // C/D layout: VGPR v holds row (lhalf*8 + v), col = lmod (within tile).
    {
        const int wtok = wave * 16;
#pragma unroll
        for (int v = 0; v < 8; ++v) {
            const int r = lhalf * 8 + v;
            float* row = &lds_logits[(wtok + r) * LDS_STRIDE];
            row[lmod]      = acc0[v];
            row[16 + lmod] = acc1[v];
            row[32 + lmod] = acc2[v];
            row[48 + lmod] = acc3[v];
        }
    }
    __syncthreads();

    // ---------------- per-token softmax / top-2 / loss partials ------------
    if (tid < 64) {
        const float* row = &lds_logits[tid * LDS_STRIDE];
        const float inv_t = 1.0f / TEMP;

        float l[NUM_EXPERTS];
        float mx = -3.4e38f;
#pragma unroll
        for (int e = 0; e < NUM_EXPERTS; ++e) {
            float v = (row[e] + bias[e]) * inv_t;
            l[e] = v;
            mx = fmaxf(mx, v);
        }
        float sum = 0.0f;
        float b0 = -3.4e38f, b1 = -3.4e38f;
        int   i0 = 0, i1 = 0;
#pragma unroll
        for (int e = 0; e < NUM_EXPERTS; ++e) {
            sum += __expf(l[e] - mx);
            if (l[e] > b0)      { b1 = b0; i1 = i0; b0 = l[e]; i0 = e; }
            else if (l[e] > b1) { b1 = l[e]; i1 = e; }
        }
        const float inv_sum = 1.0f / sum;
        const float p0 = __expf(b0 - mx) * inv_sum;
        const float p1 = __expf(b1 - mx) * inv_sum;
        const float inv_top = 1.0f / (p0 + p1);

        const int gt = wgTok0 + tid;
        out_scores[gt * 2 + 0] = p0 * inv_top;
        out_scores[gt * 2 + 1] = p1 * inv_top;
        out_idx[gt * 2 + 0] = i0;
        out_idx[gt * 2 + 1] = i1;

        const float lse = __logf(sum) + mx;
        atomicAdd(&lds_z, lse * lse);
        atomicAdd(&lds_sel[i0], 1.0f);
        atomicAdd(&lds_sel[i1], 1.0f);
#pragma unroll
        for (int e = 0; e < NUM_EXPERTS; ++e)
            atomicAdd(&lds_prob[e], __expf(l[e] - mx) * inv_sum);
    }
    __syncthreads();

    // flush block partials to global workspace
    if (tid < NUM_EXPERTS) {
        atomicAdd(&ws[tid], lds_sel[tid]);
        atomicAdd(&ws[NUM_EXPERTS + tid], lds_prob[tid]);
    }
    if (tid == 0) atomicAdd(&ws[128], lds_z);
}

// ---------------------------------------------------------------------------
// Combine reductions into the scalar loss.
// ---------------------------------------------------------------------------
__global__ void finalize_kernel(const float* __restrict__ ws,
                                float* __restrict__ out_loss, int T) {
    if (threadIdx.x == 0 && blockIdx.x == 0) {
        const float invT = 1.0f / (float)T;
        float dot = 0.0f;
        for (int e = 0; e < NUM_EXPERTS; ++e)
            dot += (ws[e] * invT) * (ws[NUM_EXPERTS + e] * invT);
        float loss = LB_ALPHA * (float)NUM_EXPERTS * dot + Z_ALPHA * (ws[128] * invT);
        *out_loss = loss;
    }
}

extern "C" void kernel_launch(void* const* d_in, const int* in_sizes, int n_in,
                              void* d_out, int out_size, void* d_ws, size_t ws_size,
                              hipStream_t stream) {
    const float* x    = (const float*)d_in[0];   // [B*S, H]
    const float* gw   = (const float*)d_in[1];   // [E, H]
    const float* bias = (const float*)d_in[2];   // [E]

    const int T = in_sizes[0] / HIDDEN;          // 16384 tokens
    const int EH = in_sizes[1];                  // E * H elements

    float* out        = (float*)d_out;
    float* out_scores = out;                              // T*2 floats
    int*   out_idx    = (int*)(out + (size_t)T * 2);      // T*2 int32
    float* out_loss   = out + (size_t)T * 4;              // 1 float

    float*    ws  = (float*)d_ws;
    _Float16* gwh = (_Float16*)((char*)d_ws + GWH_OFFSET);

    zero_ws_kernel<<<1, 256, 0, stream>>>(ws);
    convert_gw_kernel<<<EH / (256 * 4), 256, 0, stream>>>(gw, gwh);
    moe_gate_kernel<<<T / 64, 128, 0, stream>>>(x, gwh, bias, out_scores, out_idx, ws);
    finalize_kernel<<<1, 64, 0, stream>>>(ws, out_loss, T);
}
